// DMoELayer_58858231824702
// MI455X (gfx1250) — matrix operand order repaired and checked by reading, not compiled, not run on hardware
//
#include <hip/hip_runtime.h>

// Problem constants (match reference)
#define B_ROWS   32768
#define D_IN     256
#define H_DIM    128
#define NEXPERT  12            // 4 shared + 4 task0 + 4 task1
#define NFRAG_EXPERT 64        // 8 col-tiles * 8 k-tiles per expert
#define GATE_FRAG_BASE (NEXPERT * NFRAG_EXPERT)   // 768
#define NFRAG_TOTAL (GATE_FRAG_BASE + 8)          // 776 (gate: 1 col-tile x 8 k-tiles)
#define FRAG_USHORTS 512       // 32 lanes * 16 bf16
#define M_TILE   128           // rows per block (8 waves * 16)
#define XPITCH   264           // 256 + 8 pad ushorts -> 528B row pitch (bank-friendly)

typedef __attribute__((ext_vector_type(16))) __bf16        v16bf;
typedef __attribute__((ext_vector_type(8)))  float         v8f;
typedef __attribute__((ext_vector_type(4)))  unsigned int  u32x4;

union FragBits { v16bf v; u32x4 u[2]; };

__device__ __forceinline__ unsigned short f32_to_bf16(float f) {
  unsigned int u = __float_as_uint(f);
  u += 0x7FFFu + ((u >> 16) & 1u);          // round-to-nearest-even
  return (unsigned short)(u >> 16);
}
__device__ __forceinline__ float bf16_to_f32(unsigned short h) {
  return __uint_as_float(((unsigned int)h) << 16);
}

__device__ __forceinline__ v16bf ld_frag2(const unsigned short* p0,
                                          const unsigned short* p1) {
  FragBits f;
  f.u[0] = *(const u32x4*)p0;   // 16B
  f.u[1] = *(const u32x4*)p1;   // 16B
  return f.v;
}
__device__ __forceinline__ v16bf ld_frag(const unsigned short* p) {
  return ld_frag2(p, p + 8);
}

__device__ __forceinline__ v8f wmma_bf16(v16bf a, v16bf b, v8f c) {
  // (neg_a, A, neg_b, B, c_mod, C, reuse_a, reuse_b)
  return __builtin_amdgcn_wmma_f32_16x16x32_bf16(false, a, false, b,
                                                 (short)0, c, false, false);
}

// ---------------------------------------------------------------------------
// Kernel 1: pack all weights (12 experts + 2 gate heads) into bf16 hi/lo
// B-fragments for v_wmma_f32_16x16x32_bf16 in contiguous k order (A is loaded the same way):
//   frag(e,n0,k0): lane l<16  -> column n0*16+l,    K = k0*32 + 0..15
//                  lane l>=16 -> column n0*16+l-16, K = k0*32 + 16..31
// per-lane 16 bf16 stored contiguously (one 32B read in the main kernel).
// ---------------------------------------------------------------------------
__global__ __launch_bounds__(512)
void prep_weights(const float* __restrict__ Wsh,   // [4,256,128]
                  const float* __restrict__ Wtk,   // [2,4,256,128]
                  const float* __restrict__ Wg,    // [2,256,8]
                  unsigned short* __restrict__ wsH,
                  unsigned short* __restrict__ wsL) {
  const int frag = blockIdx.x;        // 0..775
  const int tid  = threadIdx.x;       // 0..511
  const int l = tid >> 4;             // lane 0..31
  const int j = tid & 15;             // element within lane
  int e, n0, k0;
  if (frag < GATE_FRAG_BASE) { e = frag >> 6; n0 = (frag >> 3) & 7; k0 = frag & 7; }
  else                       { e = NEXPERT;   n0 = 0;               k0 = frag - GATE_FRAG_BASE; }
  const int n = n0 * 16 + (l & 15);
  const int k = k0 * 32 + ((l & 16) ? 16 : 0) + j;

  float w;
  if (e < 4)            w = Wsh[((size_t)e * D_IN + k) * H_DIM + n];
  else if (e < NEXPERT) w = Wtk[((size_t)(e - 4) * D_IN + k) * H_DIM + n];
  else { int t = n >> 3, g = n & 7;  w = Wg[((size_t)t * D_IN + k) * 8 + g]; }

  const unsigned short hi = f32_to_bf16(w);
  const unsigned short lo = f32_to_bf16(w - bf16_to_f32(hi));
  const size_t off = (size_t)frag * FRAG_USHORTS + tid;
  wsH[off] = hi;
  wsL[off] = lo;
}

// ---------------------------------------------------------------------------
// Kernel 2: fused DMoE forward. 256 threads = 8 waves; each wave owns 16 rows.
// ---------------------------------------------------------------------------
__global__ __launch_bounds__(256)
void moe_main(const float* __restrict__ x,
              const unsigned short* __restrict__ wsH,
              const unsigned short* __restrict__ wsL,
              const float* __restrict__ b_share,   // [4,128]
              const float* __restrict__ b_task,    // [2,4,128]
              const float* __restrict__ b_gate,    // [2,8]
              float* __restrict__ out) {           // [2,B,128]
  __shared__ unsigned short xhi[M_TILE * XPITCH];  // x tile, bf16 hi
  __shared__ unsigned short xlo[M_TILE * XPITCH];  // x tile, bf16 lo
  __shared__ float glds[M_TILE * 16];              // gate logits -> softmax weights
  __shared__ float blds[NEXPERT * H_DIM + 16];     // expert biases + gate bias

  const int tid   = threadIdx.x;
  const int lane  = tid & 31;
  const int wbase = (tid >> 5) * 16;               // first row of this wave's tile
  const int mbase = blockIdx.x * M_TILE;

  // ---- stage x tile: f32 -> split bf16 hi/lo into LDS (coalesced b128) ----
  const float4* xsrc = (const float4*)(x + (size_t)mbase * D_IN);
  #pragma unroll
  for (int i = 0; i < (M_TILE * D_IN / 4) / 256; ++i) {
    const int idx4 = tid + i * 256;
    const int row  = idx4 >> 6;                    // 64 float4 per row
    const int col  = (idx4 & 63) << 2;
    const float4 v = xsrc[idx4];
    unsigned short* ph = &xhi[row * XPITCH + col];
    unsigned short* pl = &xlo[row * XPITCH + col];
    const float fv[4] = {v.x, v.y, v.z, v.w};
    #pragma unroll
    for (int c = 0; c < 4; ++c) {
      const unsigned short h = f32_to_bf16(fv[c]);
      ph[c] = h;
      pl[c] = f32_to_bf16(fv[c] - bf16_to_f32(h));
    }
  }
  // ---- stage biases ----
  for (int idx = tid; idx < NEXPERT * H_DIM + 16; idx += 256) {
    float bv;
    if (idx < 512)       bv = b_share[idx];
    else if (idx < 1536) bv = b_task[idx - 512];
    else                 bv = b_gate[idx - 1536];
    blds[idx] = bv;
  }
  __syncthreads();

  const int arow  = wbase + (lane & 15);
  const int asel  = (lane & 16) ? 16 : 0;          // A-fragment K-subtile select
  const int rbase = wbase + ((lane & 16) ? 8 : 0); // C-layout row base per lane half

  // ---- gate logits via WMMA (both tasks in one 16-col tile) ----
  {
    v8f acc = {0.f, 0.f, 0.f, 0.f, 0.f, 0.f, 0.f, 0.f};
    #pragma unroll
    for (int k0 = 0; k0 < 8; ++k0) {
      const int ca = k0 * 32 + asel;
      const v16bf aH = ld_frag2(&xhi[arow * XPITCH + ca], &xhi[arow * XPITCH + ca + 8]);
      const v16bf aL = ld_frag2(&xlo[arow * XPITCH + ca], &xlo[arow * XPITCH + ca + 8]);
      const size_t fo = (size_t)(GATE_FRAG_BASE + k0) * FRAG_USHORTS + lane * 16;
      const v16bf bH = ld_frag(wsH + fo);
      const v16bf bL = ld_frag(wsL + fo);
      acc = wmma_bf16(aH, bH, acc);
      acc = wmma_bf16(aH, bL, acc);
      acc = wmma_bf16(aL, bH, acc);
    }
    const float gb = blds[NEXPERT * H_DIM + (lane & 15)];
    #pragma unroll
    for (int j = 0; j < 8; ++j)
      glds[(rbase + j) * 16 + (lane & 15)] = acc[j] + gb;
  }
  __syncthreads();

  // ---- softmax: one (row, task) per thread (exactly 256 items) ----
  {
    float* g = &glds[(tid >> 1) * 16 + (tid & 1) * 8];
    float m = g[0];
    #pragma unroll
    for (int i = 1; i < 8; ++i) m = fmaxf(m, g[i]);
    float e8[8], s = 0.f;
    #pragma unroll
    for (int i = 0; i < 8; ++i) { e8[i] = __expf(g[i] - m); s += e8[i]; }
    const float inv = 1.f / s;
    #pragma unroll
    for (int i = 0; i < 8; ++i) g[i] = e8[i] * inv;
  }
  __syncthreads();

  // ---- expert GEMMs + on-the-fly gated combine into tower accumulators ----
  v8f twA[8], twB[8];                              // task0 / task1 towers
  {
    const v8f z = {0.f, 0.f, 0.f, 0.f, 0.f, 0.f, 0.f, 0.f};
    #pragma unroll
    for (int n0 = 0; n0 < 8; ++n0) { twA[n0] = z; twB[n0] = z; }
  }

  #pragma unroll 1
  for (int e = 0; e < NEXPERT; ++e) {
    // gate scalars for this expert (row-dependent only; reused across col-tiles)
    const bool has0 = (e < 8);                     // shared + task0 experts
    const bool has1 = (e < 4) || (e >= 8);         // shared + task1 experts
    const int  gi0  = e;                           // gate index, task 0
    const int  gi1  = (e < 4) ? e : (e - 4);       // gate index, task 1
    float g0[8], g1[8];
    #pragma unroll
    for (int j = 0; j < 8; ++j) {
      g0[j] = has0 ? glds[(rbase + j) * 16 + gi0]     : 0.f;
      g1[j] = has1 ? glds[(rbase + j) * 16 + 8 + gi1] : 0.f;
    }

    v8f acc[8];
    {
      const v8f z = {0.f, 0.f, 0.f, 0.f, 0.f, 0.f, 0.f, 0.f};
      #pragma unroll
      for (int n0 = 0; n0 < 8; ++n0) acc[n0] = z;
    }

    const size_t ebase = (size_t)e * NFRAG_EXPERT * FRAG_USHORTS + lane * 16;
    #pragma unroll
    for (int k0 = 0; k0 < 8; ++k0) {
      const int ca = k0 * 32 + asel;
      const v16bf aH = ld_frag2(&xhi[arow * XPITCH + ca], &xhi[arow * XPITCH + ca + 8]);
      const v16bf aL = ld_frag2(&xlo[arow * XPITCH + ca], &xlo[arow * XPITCH + ca + 8]);
      #pragma unroll
      for (int n0 = 0; n0 < 8; ++n0) {
        const size_t fo = ebase + (size_t)(n0 * 8 + k0) * FRAG_USHORTS;
        const v16bf bH = ld_frag(wsH + fo);
        const v16bf bL = ld_frag(wsL + fo);
        acc[n0] = wmma_bf16(aH, bH, acc[n0]);      // hi*hi
        acc[n0] = wmma_bf16(aH, bL, acc[n0]);      // hi*lo
        acc[n0] = wmma_bf16(aL, bH, acc[n0]);      // lo*hi  (~fp32 fidelity)
      }
    }

    // bias + ReLU + gated accumulate into towers
    #pragma unroll
    for (int n0 = 0; n0 < 8; ++n0) {
      const float bsc = blds[e * H_DIM + n0 * 16 + (lane & 15)];
      #pragma unroll
      for (int j = 0; j < 8; ++j) {
        float v = acc[n0][j] + bsc;
        v = v > 0.f ? v : 0.f;
        twA[n0][j] += g0[j] * v;
        twB[n0][j] += g1[j] * v;
      }
    }
  }

  // ---- store towers: out[t][mbase+row][h] ----
  float* o0 = out + (size_t)mbase * H_DIM;
  float* o1 = out + ((size_t)B_ROWS + mbase) * H_DIM;
  #pragma unroll
  for (int n0 = 0; n0 < 8; ++n0) {
    const int h = n0 * 16 + (lane & 15);
    #pragma unroll
    for (int j = 0; j < 8; ++j) {
      const int row = rbase + j;
      o0[(size_t)row * H_DIM + h] = twA[n0][j];
      o1[(size_t)row * H_DIM + h] = twB[n0][j];
    }
  }
}

extern "C" void kernel_launch(void* const* d_in, const int* in_sizes, int n_in,
                              void* d_out, int out_size, void* d_ws, size_t ws_size,
                              hipStream_t stream) {
  (void)in_sizes; (void)n_in; (void)out_size; (void)ws_size;
  const float* x       = (const float*)d_in[0];
  const float* W_share = (const float*)d_in[1];
  const float* b_share = (const float*)d_in[2];
  const float* W_task  = (const float*)d_in[3];
  const float* b_task  = (const float*)d_in[4];
  const float* W_gate  = (const float*)d_in[5];
  const float* b_gate  = (const float*)d_in[6];

  unsigned short* wsH = (unsigned short*)d_ws;                       // 776*512*2 B
  unsigned short* wsL = wsH + (size_t)NFRAG_TOTAL * FRAG_USHORTS;    // + same

  prep_weights<<<NFRAG_TOTAL, 512, 0, stream>>>(W_share, W_task, W_gate, wsH, wsL);
  moe_main<<<B_ROWS / M_TILE, 256, 0, stream>>>(x, wsH, wsL,
                                                b_share, b_task, b_gate,
                                                (float*)d_out);
}
